// CWS_9388798509396
// MI455X (gfx1250) — matrix operand order repaired and checked by reading, not compiled, not run on hardware
//
#include <hip/hip_runtime.h>
#include <hip/hip_bf16.h>
#include <math.h>

#define B_  64
#define T_  512
#define E_  256
#define HD_ 256
#define H_  512
#define G_  1024
#define NT_ 4

typedef __attribute__((ext_vector_type(16))) __bf16 bf16x16;
typedef __attribute__((ext_vector_type(8)))  float  f32x8;
typedef __attribute__((ext_vector_type(4)))  unsigned u32x4;
typedef __attribute__((ext_vector_type(8)))  int      i32x8;
typedef __attribute__((ext_vector_type(4)))  int      i32x4;

static __device__ __forceinline__ f32x8 wmma_bf16(bf16x16 a, bf16x16 b, f32x8 c) {
  // D = A(16x32 bf16) x B(32x16 bf16) + C(16x16 f32)
  return __builtin_amdgcn_wmma_f32_16x16x32_bf16(false, a, false, b, (short)0, c, false, false);
}

// TDM: 2D bf16 tile (tile_d0 x tile_d1 elements) global -> LDS, with optional
// LDS row padding (D# groups per CDNA5 ISA).
// pad_interval_code: 0=2,1=4,...,6=128,7=256 DWORDs; pad_amount_m1: DWORDs-1.
// This toolchain's builtin takes 6 args (extra int32x8 descriptor group).
static __device__ __forceinline__ void tdm_load_2d_bf16(
    unsigned lds_off, const __bf16* gptr,
    int tile_d0, int tile_d1, int tensor_d0, int tensor_d1, int stride_elems,
    int pad_interval_code, int pad_amount_m1) {
  unsigned long long ga = (unsigned long long)(const void*)gptr;
  u32x4 g0;
  g0[0] = 1u;                                  // count=1, user mode, no gather
  g0[1] = lds_off;                             // LDS byte address
  g0[2] = (unsigned)(ga & 0xffffffffu);        // global_addr[31:0]
  g0[3] = (unsigned)((ga >> 32) & 0x1ffffffu)  // global_addr[56:32]
        | (2u << 30);                          // type=2 ("image")
  i32x8 g1;
  const int pad_en = (pad_amount_m1 >= 0) ? 1 : 0;
  g1[0] = (1 << 16)                            // data_size code 1 = 2 bytes
        | (pad_en << 20)
        | (pad_interval_code << 22)
        | ((pad_en ? pad_amount_m1 : 0) << 25);
  g1[1] = (tensor_d0 & 0xffff) << 16;          // tensor_dim0[15:0]
  g1[2] = ((tensor_d0 >> 16) & 0xffff) | ((tensor_d1 & 0xffff) << 16);
  g1[3] = ((tensor_d1 >> 16) & 0xffff) | ((tile_d0 & 0xffff) << 16);
  g1[4] = (tile_d1 & 0xffff);                  // tile_dim1, tile_dim2=0
  g1[5] = stride_elems;                        // tensor_dim0_stride[31:0]
  g1[6] = 0;
  g1[7] = 0;
  const i32x4 z4 = {0, 0, 0, 0};
  const i32x8 z8 = {0, 0, 0, 0, 0, 0, 0, 0};
  __builtin_amdgcn_tensor_load_to_lds(g0, g1, z4, z4, z8, 0);
}

// ---------------------------------------------------------------------------
// K0: embedding gather (f32 table -> bf16 activations) + zero-init of h state
// ---------------------------------------------------------------------------
__global__ void k_embed_gather(const int* __restrict__ sentence,
                               const float* __restrict__ embedding,
                               __bf16* __restrict__ embBF,
                               __bf16* __restrict__ hcf, __bf16* __restrict__ hcb,
                               unsigned* __restrict__ bar) {
  const int bt = blockIdx.x;          // b*T + t
  const int e  = threadIdx.x;         // 0..255
  const int tok = sentence[bt];
  embBF[(size_t)bt * E_ + e] = (__bf16)embedding[(size_t)tok * E_ + e];
  if (blockIdx.x == 0) {
    for (int i = threadIdx.x; i < B_ * HD_; i += blockDim.x) {
      hcf[i] = (__bf16)0.f; hcb[i] = (__bf16)0.f;
    }
    if (threadIdx.x == 0) bar[0] = 0u;
  }
}

// ---------------------------------------------------------------------------
// K0b: cast both w_hh matrices to a contiguous bf16 copy (TDM cannot convert)
// ---------------------------------------------------------------------------
__global__ void k_cast_whh(const float* __restrict__ whhf,
                           const float* __restrict__ whhb,
                           __bf16* __restrict__ whhBF) {
  const int i = blockIdx.x * blockDim.x + threadIdx.x;   // 0 .. 2*G*HD-1
  whhBF[i] = (__bf16)((i < G_ * HD_) ? whhf[i] : whhb[i - G_ * HD_]);
}

// ---------------------------------------------------------------------------
// K1: xg = embeds @ w_ih^T + b   (both directions; M=32768, N=1024, K=256)
// Workgroup tile 128x128, 8 waves, each wave: 16(M) x 128(N), 8 accum tiles.
// ---------------------------------------------------------------------------
__global__ __launch_bounds__(256) void k_xgates(
    const __bf16* __restrict__ embBF,
    const float* __restrict__ wihf, const float* __restrict__ wihb,
    const float* __restrict__ bf,   const float* __restrict__ bb,
    __bf16* __restrict__ xgf,       __bf16* __restrict__ xgb) {
  const int dir = blockIdx.z;
  const float* wih  = dir ? wihb : wihf;
  const float* bias = dir ? bb   : bf;
  __bf16* xg        = dir ? xgb  : xgf;

  __shared__ __bf16 As[128 * 40];   // 128 rows (M) x 32 K, stride 40 (pad)
  __shared__ __bf16 Bs[128 * 40];   // 128 rows (N) x 32 K, stride 40 (pad)

  const int tid  = threadIdx.x;
  const int lane = tid & 31;
  const int w    = tid >> 5;
  const int m_blk = blockIdx.x * 128;
  const int n_blk = blockIdx.y * 128;

  f32x8 acc[8];
  for (int j = 0; j < 8; ++j)
    for (int r = 0; r < 8; ++r) acc[j][r] = 0.f;

  for (int k0 = 0; k0 < E_; k0 += 32) {
    {
      const int row = tid >> 1, cb = (tid & 1) * 16;
      const __bf16* asrc = &embBF[(size_t)(m_blk + row) * E_ + k0 + cb];
      #pragma unroll
      for (int i = 0; i < 16; ++i) As[row * 40 + cb + i] = asrc[i];
      const float* bsrc = &wih[(size_t)(n_blk + row) * E_ + k0 + cb];
      #pragma unroll
      for (int i = 0; i < 16; ++i) Bs[row * 40 + cb + i] = (__bf16)bsrc[i];
    }
    __syncthreads();

    bf16x16 a;
    {
      const int rl   = w * 16 + (lane & 15);
      const int base = (lane >> 4) * 8;
      #pragma unroll
      for (int i = 0; i < 8; ++i) {
        a[i]     = As[rl * 40 + base + i];
        a[8 + i] = As[rl * 40 + 16 + base + i];
      }
    }
    #pragma unroll
    for (int j = 0; j < 8; ++j) {
      bf16x16 bfrag;
      const int nl = j * 16 + (lane & 15);
      const int kb = (lane >> 4) * 8;
      #pragma unroll
      for (int i = 0; i < 16; ++i) bfrag[i] = Bs[nl * 40 + kb + (i & 7) + ((i >> 3) << 4)];
      acc[j] = wmma_bf16(a, bfrag, acc[j]);
    }
    __syncthreads();
  }

  #pragma unroll
  for (int j = 0; j < 8; ++j) {
    const int n = n_blk + j * 16 + (lane & 15);
    const float bv = bias[n];
    #pragma unroll
    for (int r = 0; r < 8; ++r) {
      const int m = m_blk + w * 16 + (lane >> 4) * 8 + r;
      xg[(size_t)m * G_ + n] = (__bf16)(acc[j][r] + bv);
    }
  }
}

// ---------------------------------------------------------------------------
// K2: persistent bidirectional LSTM recurrence.
// 8 blocks = 2 dirs x 4 chunks of 64 h-columns. Each block keeps its w_hh
// slice (256 gate rows x 256 K, bf16 = 128 KB) resident in LDS (320KB/WGP);
// the slice preload and the per-step h broadcast (64x256 bf16 from L2) are
// Tensor Data Mover DMAs with LDS row-padding to the 264-element stride.
// Grid-wide step sync via monotonic atomic barrier + s_sleep spin.
// ---------------------------------------------------------------------------
#define WHH_BYTES (256 * 264 * 2)
#define HSH_BYTES (64 * 264 * 2)

__global__ __launch_bounds__(256) void k_lstm(
    const __bf16* __restrict__ xgf, const __bf16* __restrict__ xgb,
    const __bf16* __restrict__ whhBF,
    const int*  __restrict__ mask,
    __bf16* __restrict__ hcf, __bf16* __restrict__ hcb,
    __bf16* __restrict__ hsf, __bf16* __restrict__ hsb,
    unsigned* __restrict__ bar) {
  const int dir = blockIdx.x >> 2;
  const int c0  = (blockIdx.x & 3) * 64;
  const __bf16* xg  = dir ? xgb  : xgf;
  const __bf16* whh = whhBF + (size_t)dir * G_ * HD_;
  __bf16* hcur      = dir ? hcb  : hcf;
  __bf16* hs        = dir ? hsb  : hsf;

  extern __shared__ char smem[];
  __bf16* whh_s = (__bf16*)smem;                                   // 256 x 264
  __bf16* h_sh  = (__bf16*)(smem + WHH_BYTES);                     // 64 x 264
  float*  gates = (float*) (smem + WHH_BYTES + HSH_BYTES);         // 64 x 260
  float*  c_sh  = (float*) (smem + WHH_BYTES + HSH_BYTES + 64 * 260 * 4);

  const int tid = threadIdx.x, lane = tid & 31, w = tid >> 5;
  const int mtile = w & 3, nhalf = w >> 2;

  // TDM preload of owned w_hh slice: rows {q*256+c0 .. +64} -> LDS rows q*64..,
  // row = 256 bf16 (128 DWORDs, pad code 6) + 4 DWORDs pad (code 3) = 264 elems.
  if (w == 0) {
    #pragma unroll
    for (int q = 0; q < 4; ++q)
      tdm_load_2d_bf16(q * 64 * 264 * 2,
                       whh + (size_t)(q * 256 + c0) * HD_,
                       /*tile*/256, 64, /*tensor*/256, 64, /*stride*/256,
                       /*pad_interval*/6, /*pad_amount-1*/3);
    __builtin_amdgcn_s_wait_tensorcnt((short)0);
  }
  for (int idx = tid; idx < 64 * 64; idx += 256) c_sh[idx] = 0.f;
  __syncthreads();

  for (int t = 0; t < T_; ++t) {
    const int tt = dir ? (T_ - 1 - t) : t;

    // TDM stage of full h_prev (64 x 256 bf16) into padded LDS tile.
    if (w == 0) {
      tdm_load_2d_bf16(WHH_BYTES, hcur,
                       256, 64, 256, 64, 256, 6, 3);
      __builtin_amdgcn_s_wait_tensorcnt((short)0);
    }
    __syncthreads();

    f32x8 acc[8];
    for (int j = 0; j < 8; ++j)
      for (int r = 0; r < 8; ++r) acc[j][r] = 0.f;

    for (int k0 = 0; k0 < HD_; k0 += 32) {
      bf16x16 a;
      const int rl   = mtile * 16 + (lane & 15);
      const int base = (lane >> 4) * 8;
      #pragma unroll
      for (int i = 0; i < 8; ++i) {
        a[i]     = h_sh[rl * 264 + k0 + base + i];
        a[8 + i] = h_sh[rl * 264 + k0 + 16 + base + i];
      }
      #pragma unroll
      for (int j = 0; j < 8; ++j) {
        bf16x16 bfrag;
        const int nl = nhalf * 128 + j * 16 + (lane & 15);
        const int kb = k0 + (lane >> 4) * 8;
        #pragma unroll
        for (int i = 0; i < 16; ++i) bfrag[i] = whh_s[nl * 264 + kb + (i & 7) + ((i >> 3) << 4)];
        acc[j] = wmma_bf16(a, bfrag, acc[j]);
      }
    }

    // gates = wmma + precomputed xg (bias already folded into xg)
    #pragma unroll
    for (int j = 0; j < 8; ++j) {
      const int n  = nhalf * 128 + j * 16 + (lane & 15);
      const int gg = (n >> 6) * 256 + c0 + (n & 63);
      #pragma unroll
      for (int r = 0; r < 8; ++r) {
        const int m = mtile * 16 + (lane >> 4) * 8 + r;  // batch index
        gates[m * 260 + n] =
            acc[j][r] + (float)xg[((size_t)m * T_ + tt) * G_ + gg];
      }
    }
    __syncthreads();

    // Pointwise LSTM cell for owned 64 h-columns.
    for (int idx = tid; idx < B_ * 64; idx += 256) {
      const int b = idx >> 6, jj = idx & 63;
      const float ig = 1.f / (1.f + __expf(-gates[b * 260 + jj]));
      const float fg = 1.f / (1.f + __expf(-gates[b * 260 + 64 + jj]));
      const float gv = tanhf(gates[b * 260 + 128 + jj]);
      const float og = 1.f / (1.f + __expf(-gates[b * 260 + 192 + jj]));
      const float co = c_sh[idx];
      const float cn = fg * co + ig * gv;
      const float hn = og * tanhf(cn);
      const int   mm = mask[b * T_ + tt];
      const float ho = (float)h_sh[b * 264 + c0 + jj];
      c_sh[idx] = mm ? cn : co;
      hcur[b * HD_ + c0 + jj] = (__bf16)(mm ? hn : ho);
      hs[((size_t)b * T_ + tt) * HD_ + c0 + jj] = (__bf16)(mm ? hn : 0.f);
    }
    __syncthreads();

    // Device-scope release + monotonic grid barrier (8 co-resident blocks).
    if (tid == 0) {
      __threadfence();
      atomicAdd(bar, 1u);
      const unsigned target = 8u * (unsigned)(t + 1);
      while (__hip_atomic_load(bar, __ATOMIC_ACQUIRE, __HIP_MEMORY_SCOPE_AGENT) < target)
        __builtin_amdgcn_s_sleep(2);
    }
    __syncthreads();
  }
}

// ---------------------------------------------------------------------------
// K3: feats[bt,k] = concat(hf,hb)[bt,:] . w_out[k,:] + b_out[k]   (N=4, tiny)
// ---------------------------------------------------------------------------
__global__ __launch_bounds__(256) void k_feats(
    const __bf16* __restrict__ hsf, const __bf16* __restrict__ hsb,
    const float* __restrict__ wout, const float* __restrict__ bout,
    float* __restrict__ feats) {
  __shared__ float ws[NT_ * H_];
  for (int i = threadIdx.x; i < NT_ * H_; i += blockDim.x) ws[i] = wout[i];
  __syncthreads();
  const int bt = blockIdx.x * blockDim.x + threadIdx.x;
  __builtin_prefetch(&hsf[(size_t)bt * HD_], 0, 1);   // global_prefetch_b8
  __builtin_prefetch(&hsb[(size_t)bt * HD_], 0, 1);
  float a0 = bout[0], a1 = bout[1], a2 = bout[2], a3 = bout[3];
  for (int j = 0; j < HD_; ++j) {
    const float hv = (float)hsf[(size_t)bt * HD_ + j];
    a0 += hv * ws[0 * H_ + j]; a1 += hv * ws[1 * H_ + j];
    a2 += hv * ws[2 * H_ + j]; a3 += hv * ws[3 * H_ + j];
  }
  for (int j = 0; j < HD_; ++j) {
    const float hv = (float)hsb[(size_t)bt * HD_ + j];
    a0 += hv * ws[0 * H_ + HD_ + j]; a1 += hv * ws[1 * H_ + HD_ + j];
    a2 += hv * ws[2 * H_ + HD_ + j]; a3 += hv * ws[3 * H_ + HD_ + j];
  }
  feats[(size_t)bt * 4 + 0] = a0; feats[(size_t)bt * 4 + 1] = a1;
  feats[(size_t)bt * 4 + 2] = a2; feats[(size_t)bt * 4 + 3] = a3;
}

// ---------------------------------------------------------------------------
// K4: CRF numerator + forward algorithm (NT=4), one thread per sequence.
// ---------------------------------------------------------------------------
__global__ void k_crf(const float* __restrict__ feats, const int* __restrict__ tags,
                      const int* __restrict__ mask, const int* __restrict__ length,
                      const float* __restrict__ start_t, const float* __restrict__ end_t,
                      const float* __restrict__ trans, float* __restrict__ out) {
  __shared__ float red[B_];
  const int b = threadIdx.x;
  float tr[NT_ * NT_];
  for (int i = 0; i < NT_ * NT_; ++i) tr[i] = trans[i];

  const float* f0 = &feats[(size_t)b * T_ * NT_];
  const int tg0 = tags[b * T_];
  float num = start_t[tg0] + f0[tg0];
  float alpha[NT_];
  for (int k = 0; k < NT_; ++k) alpha[k] = start_t[k] + f0[k];

  for (int t = 1; t < T_; ++t) {
    const float* ft = f0 + t * NT_;
    const int mm = mask[b * T_ + t];
    const int tg = tags[b * T_ + t];
    const int tp = tags[b * T_ + t - 1];
    num += mm ? (ft[tg] + tr[tp * NT_ + tg]) : 0.f;
    float na[NT_];
    #pragma unroll
    for (int k = 0; k < NT_; ++k) {
      const float m0 = alpha[0] + tr[0 * NT_ + k];
      const float m1 = alpha[1] + tr[1 * NT_ + k];
      const float m2 = alpha[2] + tr[2 * NT_ + k];
      const float m3 = alpha[3] + tr[3 * NT_ + k];
      const float mx = fmaxf(fmaxf(m0, m1), fmaxf(m2, m3));
      const float s = __expf(m0 - mx) + __expf(m1 - mx) + __expf(m2 - mx) + __expf(m3 - mx);
      na[k] = mx + __logf(s) + ft[k];
    }
    #pragma unroll
    for (int k = 0; k < NT_; ++k) alpha[k] = mm ? na[k] : alpha[k];
  }
  num += end_t[tags[b * T_ + length[b] - 1]];
  const float m0 = alpha[0] + end_t[0], m1 = alpha[1] + end_t[1];
  const float m2 = alpha[2] + end_t[2], m3 = alpha[3] + end_t[3];
  const float mx = fmaxf(fmaxf(m0, m1), fmaxf(m2, m3));
  const float logZ = mx + __logf(__expf(m0 - mx) + __expf(m1 - mx) +
                                 __expf(m2 - mx) + __expf(m3 - mx));
  red[b] = num - logZ;
  __syncthreads();
  if (b == 0) {
    float s = 0.f;
    for (int i = 0; i < B_; ++i) s += red[i];
    out[0] = -s / (float)B_;
  }
}

// ---------------------------------------------------------------------------
extern "C" void kernel_launch(void* const* d_in, const int* in_sizes, int n_in,
                              void* d_out, int out_size, void* d_ws, size_t ws_size,
                              hipStream_t stream) {
  (void)in_sizes; (void)n_in; (void)out_size; (void)ws_size;
  const int*   sentence  = (const int*)  d_in[0];
  const int*   tags      = (const int*)  d_in[1];
  const int*   mask      = (const int*)  d_in[2];
  const int*   length    = (const int*)  d_in[3];
  const float* embedding = (const float*)d_in[4];
  const float* w_ih_f    = (const float*)d_in[5];
  const float* w_hh_f    = (const float*)d_in[6];
  const float* b_f       = (const float*)d_in[7];
  const float* w_ih_b    = (const float*)d_in[8];
  const float* w_hh_b    = (const float*)d_in[9];
  const float* b_b       = (const float*)d_in[10];
  const float* w_out     = (const float*)d_in[11];
  const float* b_out     = (const float*)d_in[12];
  const float* start_t   = (const float*)d_in[13];
  const float* end_t     = (const float*)d_in[14];
  const float* trans     = (const float*)d_in[15];

  char* ws = (char*)d_ws;
  size_t off = 0;
  auto alloc = [&](size_t bytes) {
    void* p = ws + off;
    off = (off + bytes + 255) & ~(size_t)255;
    return p;
  };
  __bf16*   embBF = (__bf16*)  alloc((size_t)B_ * T_ * E_ * 2);
  __bf16*   xgf   = (__bf16*)  alloc((size_t)B_ * T_ * G_ * 2);
  __bf16*   xgb   = (__bf16*)  alloc((size_t)B_ * T_ * G_ * 2);
  __bf16*   hsf   = (__bf16*)  alloc((size_t)B_ * T_ * HD_ * 2);
  __bf16*   hsb   = (__bf16*)  alloc((size_t)B_ * T_ * HD_ * 2);
  __bf16*   hcf   = (__bf16*)  alloc((size_t)B_ * HD_ * 2);
  __bf16*   hcb   = (__bf16*)  alloc((size_t)B_ * HD_ * 2);
  __bf16*   whhBF = (__bf16*)  alloc((size_t)2 * G_ * HD_ * 2);
  float*    feats = (float*)   alloc((size_t)B_ * T_ * NT_ * 4);
  unsigned* bar   = (unsigned*)alloc(256);

  hipLaunchKernelGGL(k_embed_gather, dim3(B_ * T_), dim3(E_), 0, stream,
                     sentence, embedding, embBF, hcf, hcb, bar);

  hipLaunchKernelGGL(k_cast_whh, dim3((2 * G_ * HD_) / 256), dim3(256), 0, stream,
                     w_hh_f, w_hh_b, whhBF);

  hipLaunchKernelGGL(k_xgates, dim3((B_ * T_) / 128, G_ / 128, 2), dim3(256), 0, stream,
                     embBF, w_ih_f, w_ih_b, b_f, b_b, xgf, xgb);

  const size_t lds = WHH_BYTES + HSH_BYTES + 64 * 260 * 4 + 64 * 64 * 4;
  hipLaunchKernelGGL(k_lstm, dim3(8), dim3(256), lds, stream,
                     xgf, xgb, whhBF, mask, hcf, hcb, hsf, hsb, bar);

  hipLaunchKernelGGL(k_feats, dim3((B_ * T_) / 256), dim3(256), 0, stream,
                     hsf, hsb, w_out, b_out, feats);

  hipLaunchKernelGGL(k_crf, dim3(1), dim3(B_), 0, stream,
                     feats, tags, mask, length, start_t, end_t, trans, (float*)d_out);
}
